// CausalSelfAttention_45681272160806
// MI455X (gfx1250) — compile-verified, matrix operand order checked by reading, not run on hardware
//
#include <hip/hip_runtime.h>

typedef __attribute__((ext_vector_type(16))) __bf16 v16bf;
typedef __attribute__((ext_vector_type(8)))  float  v8f;
typedef int v4i __attribute__((vector_size(16)));   // native clang vector (what the async builtin wants)

#if defined(__has_builtin)
#if __has_builtin(__builtin_amdgcn_global_load_async_to_lds_b128)
#define HAVE_ASYNC_LDS 1
#endif
#endif
#ifndef HAVE_ASYNC_LDS
#define HAVE_ASYNC_LDS 0
#endif

typedef __attribute__((address_space(1))) v4i gv4i;
typedef __attribute__((address_space(3))) v4i lv4i;

union Frag {
    int4           i2[2];
    v16bf          v;
    unsigned short u[16];
};

__device__ __forceinline__ unsigned short f2bf(float f) {
    union { float f; unsigned u; } x; x.f = f;
    unsigned r = x.u + 0x7FFFu + ((x.u >> 16) & 1u);
    return (unsigned short)(r >> 16);
}

__device__ __forceinline__ v8f wmma_bf16(v16bf a, v16bf b, v8f c) {
    return __builtin_amdgcn_wmma_f32_16x16x32_bf16(false, a, false, b, (short)0, c, false, false);
}

// async memory -> LDS copy of 16 bytes (no VGPR staging), ASYNCcnt-tracked
__device__ __forceinline__ void cp_b128(const void* g, void* l) {
#if HAVE_ASYNC_LDS
    __builtin_amdgcn_global_load_async_to_lds_b128(
        (gv4i*)(void*)(unsigned long long)g,
        (lv4i*)l, 0, 0);
#else
    *(int4*)l = *(const int4*)g;
#endif
}

__device__ __forceinline__ void wait_async() {
#if HAVE_ASYNC_LDS
    asm volatile("s_wait_asynccnt 0x0" ::: "memory");
#endif
}

// ---------------- conversion kernels ----------------
__global__ __launch_bounds__(256) void cvt_bf16(const float* __restrict__ in,
                                                unsigned short* __restrict__ out, int n) {
    for (int i = blockIdx.x * blockDim.x + threadIdx.x; i < n; i += gridDim.x * blockDim.x)
        out[i] = f2bf(in[i]);
}

// in: K x N (row-major, K==1024), out: N x K bf16 (transposed, coalesced writes)
__global__ __launch_bounds__(256) void cvtT_bf16(const float* __restrict__ in,
                                                 unsigned short* __restrict__ out, int N) {
    int total = N << 10;
    for (int i = blockIdx.x * blockDim.x + threadIdx.x; i < total; i += gridDim.x * blockDim.x) {
        int n = i >> 10, k = i & 1023;
        out[i] = f2bf(in[k * N + n]);
    }
}

// ---------------- tiled bf16 GEMM (double-buffered, async-to-LDS) ----------------
// C[M x Nn] = A[M x 1024] * Bt[Nn x 1024]^T + bias
// mode 0: scatter bf16 into Q/K/V [B,H,S,64]   mode 1: f32 row-major into Fo
__global__ __launch_bounds__(256) void gemm128(const unsigned short* __restrict__ A,
                                               const unsigned short* __restrict__ Bt,
                                               const float* __restrict__ bias,
                                               unsigned short* __restrict__ Qo,
                                               unsigned short* __restrict__ Ko,
                                               unsigned short* __restrict__ Vo,
                                               float* __restrict__ Fo,
                                               int mode) {
    const int K = 1024;
    __shared__ unsigned short sA[2][128 * 48];
    __shared__ unsigned short sB[2][128 * 48];

    const int tid   = threadIdx.x;
    const int lane  = tid & 31, wave = tid >> 5;
    const int waveM = wave >> 2, waveN = wave & 3;
    const int lrow  = lane & 15, hi = lane >> 4;
    const int m0 = blockIdx.x * 128, n0 = blockIdx.y * 128;

    v8f acc[4][2];
    const v8f vzero = {0.f, 0.f, 0.f, 0.f, 0.f, 0.f, 0.f, 0.f};
    for (int i = 0; i < 4; ++i)
        for (int t = 0; t < 2; ++t) acc[i][t] = vzero;

    const int arow = tid >> 1;          // 0..127
    const int aseg = (tid & 1) * 16;    // element offset within 32-wide K tile

    const unsigned short* gA = A  + (size_t)(m0 + arow) * K + aseg;
    const unsigned short* gB = Bt + (size_t)(n0 + arow) * K + aseg;
    unsigned short* lA0 = &sA[0][arow * 48 + aseg];
    unsigned short* lB0 = &sB[0][arow * 48 + aseg];
    unsigned short* lA1 = &sA[1][arow * 48 + aseg];
    unsigned short* lB1 = &sB[1][arow * 48 + aseg];

    // prologue: stage 0
    cp_b128(gA, lA0);       cp_b128(gA + 8, lA0 + 8);
    cp_b128(gB, lB0);       cp_b128(gB + 8, lB0 + 8);

    for (int kt = 0; kt < 32; ++kt) {
        const int buf = kt & 1;
        wait_async();
        __syncthreads();
        if (kt < 31) {  // prefetch stage kt+1 into the other buffer
            const int k1 = (kt + 1) * 32;
            unsigned short* dA = (buf ? lA0 : lA1);
            unsigned short* dB = (buf ? lB0 : lB1);
            cp_b128(gA + k1, dA);       cp_b128(gA + k1 + 8, dA + 8);
            cp_b128(gB + k1, dB);       cp_b128(gB + k1 + 8, dB + 8);
        }

        Frag af[4], bfr[2];
#pragma unroll
        for (int i = 0; i < 4; ++i) {
            const unsigned short* p = &sA[buf][(waveM * 64 + i * 16 + lrow) * 48 + hi * 8];
            af[i].i2[0] = *(const int4*)p;
            af[i].i2[1] = *(const int4*)(p + 16);
        }
#pragma unroll
        for (int t = 0; t < 2; ++t) {
            const unsigned short* p = &sB[buf][(waveN * 32 + t * 16 + lrow) * 48 + hi * 8];
            bfr[t].i2[0] = *(const int4*)p;
            bfr[t].i2[1] = *(const int4*)(p + 16);
        }
#pragma unroll
        for (int i = 0; i < 4; ++i)
#pragma unroll
            for (int t = 0; t < 2; ++t)
                acc[i][t] = wmma_bf16(af[i].v, bfr[t].v, acc[i][t]);
    }

    // epilogue: C layout — lane col = lrow, rows = v + hi*8 ; tile-uniform scalars hoisted
#pragma unroll
    for (int t = 0; t < 2; ++t) {
        const int gnBase = n0 + waveN * 32 + t * 16;         // uniform in wave
        const float bv = bias[gnBase + lrow];
        if (mode == 0) {
            const int sel = gnBase >> 10;                    // 0:q 1:k 2:v (uniform)
            unsigned short* P = (sel == 0) ? Qo : (sel == 1) ? Ko : Vo;
            const int colB = gnBase & 1023;
            const int h = colB >> 6;                         // head (uniform: 16-col tile never crosses 64)
            const int d = (colB & 63) + lrow;
#pragma unroll
            for (int i = 0; i < 4; ++i)
#pragma unroll
                for (int v = 0; v < 8; ++v) {
                    const int gm = m0 + waveM * 64 + i * 16 + v + hi * 8;
                    const int b = gm >> 11, s = gm & 2047;
                    P[((((size_t)b * 16 + h) * 2048 + s) * 64) + d] = f2bf(acc[i][t][v] + bv);
                }
        } else {
#pragma unroll
            for (int i = 0; i < 4; ++i)
#pragma unroll
                for (int v = 0; v < 8; ++v) {
                    const int gm = m0 + waveM * 64 + i * 16 + v + hi * 8;
                    Fo[(size_t)gm * 1024 + gnBase + lrow] = acc[i][t][v] + bv;
                }
        }
    }
}

// ---------------- flash attention: block = 128 q rows, wave = 16 q rows ----------------
// K tile async-staged row-major in LDS; V tile transposed into LDS cooperatively.
__global__ __launch_bounds__(256) void attn(const unsigned short* __restrict__ Q,
                                            const unsigned short* __restrict__ K,
                                            const unsigned short* __restrict__ V,
                                            unsigned short* __restrict__ Y) {
    __shared__ unsigned short sK[32 * 72];       // [key][hd]   stride 72
    __shared__ unsigned short sVT[64 * 48];      // [hd][key]   stride 48
    __shared__ unsigned short sP[8][16 * 48];    // per-wave P transpose

    const int tid  = threadIdx.x;
    const int lane = tid & 31, wave = tid >> 5;
    const int lrow = lane & 15, hi = lane >> 4;
    const int bh   = blockIdx.x >> 4;            // b*16 + h
    const int qblk = blockIdx.x & 15;
    const int b = bh >> 4, h = bh & 15;
    const int qbase = qblk * 128 + wave * 16;

    const unsigned short* Qp = Q + (size_t)bh * 2048 * 64;
    const unsigned short* Kp = K + (size_t)bh * 2048 * 64;
    const unsigned short* Vp = V + (size_t)bh * 2048 * 64;

    // Q fragments (16 rows x 64 hd -> 2 A-frags), kept in registers
    Frag qf[2];
#pragma unroll
    for (int g = 0; g < 2; ++g) {
        const unsigned short* pp = Qp + (size_t)(qbase + lrow) * 64 + g * 32 + hi * 8;
        qf[g].i2[0] = *(const int4*)pp;
        qf[g].i2[1] = *(const int4*)(pp + 16);
    }

    const v8f vzero = {0.f, 0.f, 0.f, 0.f, 0.f, 0.f, 0.f, 0.f};
    v8f o[4];
    float m_[8], l_[8];
#pragma unroll
    for (int g = 0; g < 4; ++g) o[g] = vzero;
#pragma unroll
    for (int v = 0; v < 8; ++v) { m_[v] = -1e30f; l_[v] = 0.f; }

    unsigned short* sp = sP[wave];
    const int jendw = ((qbase + 15) >> 5) << 5;       // this wave's last key block
    const int jendb = qblk * 128 + 96;                // block's last key block

    const int krow = tid >> 3;            // 0..31 (key within block)
    const int kseg = (tid & 7) * 8;       // 0..56 (hd segment)

    for (int j = 0; j <= jendb; j += 32) {
        // --- cooperative staging ---
        cp_b128(Kp + (size_t)(j + krow) * 64 + kseg, &sK[krow * 72 + kseg]);   // async
        int4 vv = *(const int4*)(Vp + (size_t)(j + krow) * 64 + kseg);          // 8 bf16 of V
        const unsigned short* pv = (const unsigned short*)&vv;
#pragma unroll
        for (int e = 0; e < 8; ++e)
            sVT[(kseg + e) * 48 + krow] = pv[e];                                // transpose into LDS
        wait_async();
        __syncthreads();

        if (j <= jendw) {
            // S = Q * K^T (16 x 32): two C tiles, contraction over hd
            v8f s[2]; s[0] = vzero; s[1] = vzero;
#pragma unroll
            for (int t = 0; t < 2; ++t) {
                const unsigned short* kp = &sK[(t * 16 + lrow) * 72];
#pragma unroll
                for (int g = 0; g < 2; ++g) {
                    Frag kf;
                    const unsigned short* pp = kp + g * 32 + hi * 8;
                    kf.i2[0] = *(const int4*)pp;
                    kf.i2[1] = *(const int4*)(pp + 16);
                    s[t] = wmma_bf16(qf[g].v, kf.v, s[t]);
                }
            }

            const bool diag = (j + 31 > qbase);
            float rowmax[8];
#pragma unroll
            for (int v = 0; v < 8; ++v) {
                float a0 = s[0][v] * 0.125f;     // 1/sqrt(64)
                float a1 = s[1][v] * 0.125f;
                if (diag) {
                    const int q = qbase + v + hi * 8;
                    if (j + lrow > q)      a0 = -1e30f;
                    if (j + 16 + lrow > q) a1 = -1e30f;
                }
                s[0][v] = a0; s[1][v] = a1;
                rowmax[v] = fmaxf(a0, a1);
            }
#pragma unroll
            for (int off = 1; off < 16; off <<= 1)
#pragma unroll
                for (int v = 0; v < 8; ++v)
                    rowmax[v] = fmaxf(rowmax[v], __shfl_xor(rowmax[v], off, 32));

            float rowsum[8];
#pragma unroll
            for (int v = 0; v < 8; ++v) {
                const float mnew = fmaxf(m_[v], rowmax[v]);
                const float corr = __expf(m_[v] - mnew);
                m_[v] = mnew;
                l_[v] *= corr;
#pragma unroll
                for (int g = 0; g < 4; ++g) o[g][v] *= corr;
                const float e0 = __expf(s[0][v] - mnew);
                const float e1 = __expf(s[1][v] - mnew);
                s[0][v] = e0; s[1][v] = e1;
                rowsum[v] = e0 + e1;
            }
#pragma unroll
            for (int off = 1; off < 16; off <<= 1)
#pragma unroll
                for (int v = 0; v < 8; ++v)
                    rowsum[v] += __shfl_xor(rowsum[v], off, 32);
#pragma unroll
            for (int v = 0; v < 8; ++v) l_[v] += rowsum[v];

            // transpose P through per-wave LDS: C-layout -> A-frag layout (bf16)
#pragma unroll
            for (int t = 0; t < 2; ++t)
#pragma unroll
                for (int v = 0; v < 8; ++v)
                    sp[(v + hi * 8) * 48 + t * 16 + lrow] = f2bf(s[t][v]);
            asm volatile("s_wait_dscnt 0" ::: "memory");   // LDS in-order per wave

            Frag pf;
            {
                const unsigned short* pp = sp + lrow * 48 + hi * 8;
                pf.i2[0] = *(const int4*)pp;
                pf.i2[1] = *(const int4*)(pp + 16);
            }

            // O += P * V : B-frags read contiguously from transposed V tile
#pragma unroll
            for (int g = 0; g < 4; ++g) {
                Frag vf;
                const unsigned short* pp = &sVT[(g * 16 + lrow) * 48 + hi * 8];
                vf.i2[0] = *(const int4*)pp;
                vf.i2[1] = *(const int4*)(pp + 16);
                o[g] = wmma_bf16(pf.v, vf.v, o[g]);
            }
        }
        __syncthreads();   // protect sK/sVT before next stage overwrites
    }

    // finalize + store bf16 into y [B,S,1024] for the output projection
#pragma unroll
    for (int v = 0; v < 8; ++v) {
        const float inv = 1.0f / l_[v];
#pragma unroll
        for (int g = 0; g < 4; ++g) o[g][v] *= inv;
    }
#pragma unroll
    for (int g = 0; g < 4; ++g) {
        const int d = g * 16 + lrow;
#pragma unroll
        for (int v = 0; v < 8; ++v) {
            const int q = qbase + v + hi * 8;
            Y[((size_t)(b * 2048 + q)) * 1024 + h * 64 + d] = f2bf(o[g][v]);
        }
    }
}

extern "C" void kernel_launch(void* const* d_in, const int* in_sizes, int n_in,
                              void* d_out, int out_size, void* d_ws, size_t ws_size,
                              hipStream_t stream) {
    const float* x     = (const float*)d_in[0];   // [4,2048,1024]
    const float* w_qkv = (const float*)d_in[1];   // [1024,3072]
    const float* b_qkv = (const float*)d_in[2];   // [3072]
    const float* w_o   = (const float*)d_in[3];   // [1024,1024]
    const float* b_o   = (const float*)d_in[4];   // [1024]
    float* out = (float*)d_out;

    char* ws = (char*)d_ws;
    unsigned short* xb    = (unsigned short*)(ws);                       // 16 MB
    unsigned short* wqkvT = (unsigned short*)(ws + 16777216);            //  6 MB
    unsigned short* woT   = (unsigned short*)(ws + 23068672);            //  2 MB
    unsigned short* Qb    = (unsigned short*)(ws + 25165824);            // 16 MB
    unsigned short* Kb    = (unsigned short*)(ws + 41943040);            // 16 MB
    unsigned short* Vb    = (unsigned short*)(ws + 58720256);            // 16 MB
    unsigned short* Yb    = (unsigned short*)(ws + 75497472);            // 16 MB

    cvt_bf16<<<4096, 256, 0, stream>>>(x, xb, 4 * 2048 * 1024);
    cvtT_bf16<<<4096, 256, 0, stream>>>(w_qkv, wqkvT, 3072);
    cvtT_bf16<<<2048, 256, 0, stream>>>(w_o, woT, 1024);

    // QKV projection: [8192 x 1024] * [1024 x 3072]
    gemm128<<<dim3(64, 24), 256, 0, stream>>>(xb, wqkvT, b_qkv, Qb, Kb, Vb, nullptr, 0);

    // causal flash attention: 64 (b,h) * 16 q-tiles of 128 rows
    attn<<<1024, 256, 0, stream>>>(Qb, Kb, Vb, Yb);

    // output projection: [8192 x 1024] * [1024 x 1024] -> f32
    gemm128<<<dim3(64, 8), 256, 0, stream>>>(Yb, woT, b_o, nullptr, nullptr, nullptr, out, 1);
}
